// GAT_gate_11458972746076
// MI455X (gfx1250) — matrix operand order repaired and checked by reading, not compiled, not run on hardware
//
#include <hip/hip_runtime.h>
#include <hip/hip_bf16.h>

// Problem constants (match reference)
#define BB   16
#define NN   1024
#define DD   128   // D_IN == D_OUT == 128
#define NHOPS 3

typedef __attribute__((ext_vector_type(16))) __bf16 v16bf;
typedef __attribute__((ext_vector_type(8)))  __bf16 v8bf;
typedef __attribute__((ext_vector_type(8)))  float  v8f;

__device__ __forceinline__ v8f vzero8() {
  v8f z;
#pragma unroll
  for (int i = 0; i < 8; ++i) z[i] = 0.0f;
  return z;
}

// ---- WMMA operand loaders (wave32, bf16 16x16x32) -------------------------
// A operand: 16x32 tile, source row-major, rows = M, contracting along row.
// Lane (m = lane&15, half = lane>>4): elems 0..7 = K[8h..8h+7], 8..15 = K[16+8h..].
__device__ __forceinline__ v16bf load_tile_A(const __bf16* __restrict__ src,
                                             int row0, int ld, int k0, int lane) {
  const int m = lane & 15, half = lane >> 4;
  const __bf16* p = src + (size_t)(row0 + m) * ld + k0;
  union { v16bf v; v8bf h[2]; } u;
  u.h[0] = *(const v8bf*)(p + half * 8);
  u.h[1] = *(const v8bf*)(p + 16 + half * 8);
  return u.v;
}

// B operand: 32x16 tile; column n of B = row n of the transposed row-major
// storage srcT. Lane (n = lane&15, half): elems 0..7 = K[8h..8h+7], 8..15 = K[16+8h..].
__device__ __forceinline__ v16bf load_tile_B(const __bf16* __restrict__ srcT,
                                             int col0, int ld, int k0, int lane) {
  const int n = lane & 15, half = lane >> 4;
  const __bf16* p = srcT + (size_t)(col0 + n) * ld + k0;
  union { v16bf v; v8bf h[2]; } u;
  u.h[0] = *(const v8bf*)(p + half * 8);
  u.h[1] = *(const v8bf*)(p + 16 + half * 8);
  return u.v;
}

// A operand loaded from f32 source with inline bf16 conversion (for x).
__device__ __forceinline__ v16bf load_tile_A_f32(const float* __restrict__ src,
                                                 int row0, int ld, int k0, int lane) {
  const int m = lane & 15, half = lane >> 4;
  const float* p = src + (size_t)(row0 + m) * ld + k0;
  v16bf r;
#pragma unroll
  for (int i = 0; i < 8; ++i) r[i] = (__bf16)p[half * 8 + i];
#pragma unroll
  for (int i = 0; i < 8; ++i) r[8 + i] = (__bf16)p[16 + half * 8 + i];
  return r;
}

#define WMMA_BF16(acc, a, b)                                                  \
  (acc) = __builtin_amdgcn_wmma_f32_16x16x32_bf16(false, (a), false, (b),     \
                                                  (short)0, (acc), false, false)

// ---- K0: convert W to bf16, build A^T in bf16 -----------------------------
__global__ void k0_prep(const float* __restrict__ W_w, const float* __restrict__ A,
                        __bf16* __restrict__ Wbf, __bf16* __restrict__ ATbf) {
  for (int i = blockIdx.x * blockDim.x + threadIdx.x; i < DD * DD;
       i += gridDim.x * blockDim.x) {
    Wbf[i] = (__bf16)W_w[i];
    const int d = i >> 7, e = i & 127;          // A[d,e]
    ATbf[e * DD + d] = (__bf16)A[i];            // AT[e,d]
  }
}

// ---- K1: h = x @ W^T + b ; emit h(f32), h(bf16), h^T(bf16), gh = h.gw1 ----
__global__ __launch_bounds__(32) void k1_h(const float* __restrict__ x,
                                           const __bf16* __restrict__ Wbf,
                                           const float* __restrict__ Wb,
                                           const float* __restrict__ gw1,
                                           float* __restrict__ h,
                                           __bf16* __restrict__ h_bf,
                                           __bf16* __restrict__ hT_bf,
                                           float* __restrict__ gh) {
  const int lane = threadIdx.x;
  const int tile = blockIdx.x;                  // B * (N/16) tiles
  const int b = tile >> 6;
  const int i0 = (tile & 63) * 16;
  const float* xb = x + (size_t)b * NN * DD;

  v8f acc[8];
#pragma unroll
  for (int t = 0; t < 8; ++t) acc[t] = vzero8();

#pragma unroll
  for (int k0 = 0; k0 < DD; k0 += 32) {
    v16bf a = load_tile_A_f32(xb, i0, DD, k0, lane);
#pragma unroll
    for (int t = 0; t < 8; ++t) {
      v16bf bt = load_tile_B(Wbf, t * 16, DD, k0, lane);
      WMMA_BF16(acc[t], a, bt);
    }
  }

  const int c = lane & 15, half = lane >> 4;
  float gpart[8];
#pragma unroll
  for (int r = 0; r < 8; ++r) gpart[r] = 0.0f;

#pragma unroll
  for (int t = 0; t < 8; ++t) {
#pragma unroll
    for (int r = 0; r < 8; ++r) {
      const int row = i0 + r + 8 * half;
      const int col = t * 16 + c;
      const float v = acc[t][r] + Wb[col];
      h[(size_t)b * NN * DD + (size_t)row * DD + col] = v;
      h_bf[(size_t)b * NN * DD + (size_t)row * DD + col] = (__bf16)v;
      hT_bf[(size_t)b * DD * NN + (size_t)col * NN + row] = (__bf16)v;
      gpart[r] += gw1[col] * v;
    }
  }
  // reduce across the 16 lanes sharing the same rows
#pragma unroll
  for (int m = 1; m < 16; m <<= 1)
#pragma unroll
    for (int r = 0; r < 8; ++r) gpart[r] += __shfl_xor(gpart[r], m, 32);
  if (c == 0) {
#pragma unroll
    for (int r = 0; r < 8; ++r) gh[b * NN + i0 + r + 8 * half] = gpart[r];
  }
}

// ---- K2: hA = h @ A (bf16 out) --------------------------------------------
__global__ __launch_bounds__(32) void k2_hA(const __bf16* __restrict__ h_bf,
                                            const __bf16* __restrict__ ATbf,
                                            __bf16* __restrict__ hA_bf) {
  const int lane = threadIdx.x;
  const int tile = blockIdx.x;
  const int b = tile >> 6;
  const int j0 = (tile & 63) * 16;
  const __bf16* hb = h_bf + (size_t)b * NN * DD;

  v8f acc[8];
#pragma unroll
  for (int t = 0; t < 8; ++t) acc[t] = vzero8();

#pragma unroll
  for (int k0 = 0; k0 < DD; k0 += 32) {
    v16bf a = load_tile_A(hb, j0, DD, k0, lane);
#pragma unroll
    for (int t = 0; t < 8; ++t) {
      v16bf bt = load_tile_B(ATbf, t * 16, DD, k0, lane);
      WMMA_BF16(acc[t], a, bt);
    }
  }

  const int c = lane & 15, half = lane >> 4;
#pragma unroll
  for (int t = 0; t < 8; ++t)
#pragma unroll
    for (int r = 0; r < 8; ++r) {
      const int row = j0 + r + 8 * half;
      hA_bf[(size_t)b * NN * DD + (size_t)row * DD + t * 16 + c] = (__bf16)acc[t][r];
    }
}

// ---- K3: e_masked = adj>0 ? (hA.h^T + h.hA^T) : 0 -------------------------
__global__ __launch_bounds__(32) void k3_e(const __bf16* __restrict__ hA_bf,
                                           const __bf16* __restrict__ h_bf,
                                           const float* __restrict__ adj,
                                           float* __restrict__ e) {
  const int lane = threadIdx.x;
  const int b = blockIdx.z;
  const int j0 = blockIdx.y * 16;
  const int k0 = blockIdx.x * 16;
  const __bf16* hb = h_bf + (size_t)b * NN * DD;
  const __bf16* ab = hA_bf + (size_t)b * NN * DD;

  v8f acc = vzero8();
#pragma unroll
  for (int kk = 0; kk < DD; kk += 32) {
    v16bf a1 = load_tile_A(ab, j0, DD, kk, lane);
    v16bf b1 = load_tile_B(hb, k0, DD, kk, lane);
    WMMA_BF16(acc, a1, b1);
    v16bf a2 = load_tile_A(hb, j0, DD, kk, lane);
    v16bf b2 = load_tile_B(ab, k0, DD, kk, lane);
    WMMA_BF16(acc, a2, b2);
  }

  const int c = lane & 15, half = lane >> 4;
#pragma unroll
  for (int r = 0; r < 8; ++r) {
    const int row = j0 + r + 8 * half;
    const size_t idx = (size_t)b * NN * NN + (size_t)row * NN + (k0 + c);
    e[idx] = (adj[idx] > 0.0f) ? acc[r] : 0.0f;
  }
}

// ---- K4: column (axis=1) max and exp-sum ----------------------------------
__global__ __launch_bounds__(256) void k4_reduce(const float* __restrict__ e,
                                                 float* __restrict__ cmax,
                                                 float* __restrict__ csum) {
  const int idx = blockIdx.x * 256 + threadIdx.x;   // b*N + k
  const int b = idx >> 10, k = idx & 1023;
  const float* col = e + (size_t)b * NN * NN + k;
  float m = -1e30f;
  for (int j = 0; j < NN; ++j) m = fmaxf(m, col[(size_t)j * NN]);
  float s = 0.0f;
  for (int j = 0; j < NN; ++j) s += __expf(col[(size_t)j * NN] - m);
  cmax[idx] = m;
  csum[idx] = s;
}

// ---- K5: att = adj * softmax(e_masked, axis=1) -> bf16 --------------------
__global__ __launch_bounds__(256) void k5_att(const float* __restrict__ e,
                                              const float* __restrict__ adj,
                                              const float* __restrict__ cmax,
                                              const float* __restrict__ csum,
                                              __bf16* __restrict__ att) {
  const size_t total = (size_t)BB * NN * NN;
  for (size_t idx = (size_t)blockIdx.x * 256 + threadIdx.x; idx < total;
       idx += (size_t)gridDim.x * 256) {
    const int b = (int)(idx >> 20);
    const int k = (int)(idx & 1023);
    const int ci = b * NN + k;
    float v = 0.0f;
    if (adj[idx] > 0.0f) v = __expf(e[idx] - cmax[ci]) / csum[ci];
    att[idx] = (__bf16)v;
  }
}

// ---- K6: one hop: az = relu(att @ retval); gated blend --------------------
__global__ __launch_bounds__(32) void k6_hop(const __bf16* __restrict__ att,
                                             const __bf16* __restrict__ rT_in,
                                             const float* __restrict__ h,
                                             const float* __restrict__ gh,
                                             const float* __restrict__ gw2,
                                             const float* __restrict__ gb,
                                             __bf16* __restrict__ rT_out,
                                             float* __restrict__ out_f32) {
  const int lane = threadIdx.x;
  const int tile = blockIdx.x;
  const int b = tile >> 6;
  const int i0 = (tile & 63) * 16;
  const __bf16* attb = att + (size_t)b * NN * NN;
  const __bf16* rTb = rT_in + (size_t)b * DD * NN;

  v8f acc[8];
#pragma unroll
  for (int t = 0; t < 8; ++t) acc[t] = vzero8();

  for (int k0 = 0; k0 < NN; k0 += 32) {
    if (k0 + 32 < NN)
      __builtin_prefetch(attb + (size_t)(i0 + (lane & 15)) * NN + k0 + 32, 0, 1);
    v16bf a = load_tile_A(attb, i0, NN, k0, lane);
#pragma unroll
    for (int t = 0; t < 8; ++t) {
      v16bf bt = load_tile_B(rTb, t * 16, NN, k0, lane);
      WMMA_BF16(acc[t], a, bt);
    }
  }

  const int c = lane & 15, half = lane >> 4;
  float gpart[8];
#pragma unroll
  for (int r = 0; r < 8; ++r) gpart[r] = 0.0f;

  // relu + gate partial dot (az half of gate_w)
#pragma unroll
  for (int t = 0; t < 8; ++t)
#pragma unroll
    for (int r = 0; r < 8; ++r) {
      float v = acc[t][r];
      v = v > 0.0f ? v : 0.0f;
      acc[t][r] = v;
      gpart[r] += gw2[t * 16 + c] * v;
    }
#pragma unroll
  for (int m = 1; m < 16; m <<= 1)
#pragma unroll
    for (int r = 0; r < 8; ++r) gpart[r] += __shfl_xor(gpart[r], m, 32);

  const float gbv = gb[0];
  float coeff[8];
#pragma unroll
  for (int r = 0; r < 8; ++r) {
    const int row = i0 + r + 8 * half;
    const float z = gh[b * NN + row] + gpart[r] + gbv;
    coeff[r] = 1.0f / (1.0f + __expf(-z));
  }

#pragma unroll
  for (int t = 0; t < 8; ++t)
#pragma unroll
    for (int r = 0; r < 8; ++r) {
      const int row = i0 + r + 8 * half;
      const int col = t * 16 + c;
      const float hv = h[(size_t)b * NN * DD + (size_t)row * DD + col];
      const float rv = coeff[r] * hv + (1.0f - coeff[r]) * acc[t][r];
      rT_out[(size_t)b * DD * NN + (size_t)col * NN + row] = (__bf16)rv;
      if (out_f32)
        out_f32[(size_t)b * NN * DD + (size_t)row * DD + col] = rv;
    }
}

// ---- host side ------------------------------------------------------------
extern "C" void kernel_launch(void* const* d_in, const int* in_sizes, int n_in,
                              void* d_out, int out_size, void* d_ws, size_t ws_size,
                              hipStream_t stream) {
  const float* x      = (const float*)d_in[0];
  const float* adj    = (const float*)d_in[1];
  const float* W_w    = (const float*)d_in[2];
  const float* W_b    = (const float*)d_in[3];
  const float* A      = (const float*)d_in[4];
  const float* gate_w = (const float*)d_in[5];   // [1, 2*DD]: [0..127]=h, [128..255]=az
  const float* gate_b = (const float*)d_in[6];
  float* out = (float*)d_out;

  char* ws = (char*)d_ws;
  size_t off = 0;
  auto alloc = [&](size_t bytes) { char* p = ws + off; off += (bytes + 255) & ~(size_t)255; return p; };

  float*  h_f32 = (float*) alloc((size_t)BB * NN * DD * 4);
  __bf16* h_bf  = (__bf16*)alloc((size_t)BB * NN * DD * 2);
  __bf16* hT_bf = (__bf16*)alloc((size_t)BB * DD * NN * 2);
  __bf16* hA_bf = (__bf16*)alloc((size_t)BB * NN * DD * 2);
  __bf16* Wbf   = (__bf16*)alloc((size_t)DD * DD * 2);
  __bf16* ATbf  = (__bf16*)alloc((size_t)DD * DD * 2);
  float*  gh    = (float*) alloc((size_t)BB * NN * 4);
  float*  e_m   = (float*) alloc((size_t)BB * NN * NN * 4);
  float*  cmax  = (float*) alloc((size_t)BB * NN * 4);
  float*  csum  = (float*) alloc((size_t)BB * NN * 4);
  __bf16* attb  = (__bf16*)alloc((size_t)BB * NN * NN * 2);
  __bf16* rT_a  = (__bf16*)alloc((size_t)BB * DD * NN * 2);
  __bf16* rT_b  = (__bf16*)alloc((size_t)BB * DD * NN * 2);
  (void)ws_size; (void)in_sizes; (void)n_in; (void)out_size;

  k0_prep<<<64, 256, 0, stream>>>(W_w, A, Wbf, ATbf);
  k1_h<<<BB * (NN / 16), 32, 0, stream>>>(x, Wbf, W_b, gate_w, h_f32, h_bf, hT_bf, gh);
  k2_hA<<<BB * (NN / 16), 32, 0, stream>>>(h_bf, ATbf, hA_bf);
  k3_e<<<dim3(NN / 16, NN / 16, BB), 32, 0, stream>>>(hA_bf, h_bf, adj, e_m);
  k4_reduce<<<(BB * NN) / 256, 256, 0, stream>>>(e_m, cmax, csum);
  k5_att<<<8192, 256, 0, stream>>>(e_m, adj, cmax, csum, attb);

  const float* gw2 = gate_w + DD;
  // hop 1: retval = h  (rT_in = h^T)
  k6_hop<<<BB * (NN / 16), 32, 0, stream>>>(attb, hT_bf, h_f32, gh, gw2, gate_b, rT_a, nullptr);
  // hop 2
  k6_hop<<<BB * (NN / 16), 32, 0, stream>>>(attb, rT_a, h_f32, gh, gw2, gate_b, rT_b, nullptr);
  // hop 3: final, write f32 output
  k6_hop<<<BB * (NN / 16), 32, 0, stream>>>(attb, rT_b, h_f32, gh, gw2, gate_b, rT_a, out);
}
